// GAT_7876970020920
// MI455X (gfx1250) — matrix operand order repaired and checked by reading, not compiled, not run on hardware
//
#include <hip/hip_runtime.h>
#include <stdint.h>

// ---------------------------------------------------------------------------
// GAT (2-layer) for MI455X / gfx1250, wave32 + WMMA bf16 path.
// Flash-style masked softmax: never materializes the N x N x H score tensor.
// W1 fragments staged in LDS via GLOBAL_LOAD_ASYNC_TO_LDS_B128 (ASYNCcnt).
// ---------------------------------------------------------------------------

typedef __bf16 bf16;
typedef __attribute__((ext_vector_type(16))) __bf16 v16bf;
typedef __attribute__((ext_vector_type(8)))  __bf16 v8bf;
typedef __attribute__((ext_vector_type(8)))  float  v8f;
typedef __attribute__((ext_vector_type(4)))  float  v4f;

#define NN    4096
#define INF   512
#define HEADS 8
#define HID   64
#define CLS   16
#define SLOPE 0.2f

// Workspace layout (bytes). Total ~1.46 MB.
#define OFF_W1S 0                         // bf16 [16][4][32][16] swizzled W1 B-frags
#define OFF_W2S (OFF_W1S + 65536)         // bf16 [2][32][16]     swizzled W2 B-frags
#define OFF_G1T (OFF_W2S + 2048)          // bf16 [64][4096]      g1 transposed (feature-major)
#define OFF_S1L (OFF_G1T + 524288)        // f32  [8][4096]       sl per head (head-major)
#define OFF_S1R (OFF_S1L + 131072)        // f32  [8][4096]       sr per head
#define OFF_H1  (OFF_S1R + 131072)        // bf16 [4096][64]      elu(layer1 out), row-major
#define OFF_G2T (OFF_H1  + 524288)        // bf16 [16][4096]      g2 transposed
#define OFF_S2L (OFF_G2T + 131072)        // f32  [4096]
#define OFF_S2R (OFF_S2L + 16384)         // f32  [4096]

static __device__ __forceinline__ v16bf combine16(v8bf lo, v8bf hi) {
  v16bf r;
#pragma unroll
  for (int t = 0; t < 8; ++t) { r[t] = lo[t]; r[t + 8] = hi[t]; }
  return r;
}

static __device__ __forceinline__ v8f wmma_bf16(v16bf a, v16bf b, v8f c) {
  return __builtin_amdgcn_wmma_f32_16x16x32_bf16(false, a, false, b,
                                                 (short)0, c, false, false);
}

static __device__ __forceinline__ v8f vzero8() {
  v8f z = {0.f, 0.f, 0.f, 0.f, 0.f, 0.f, 0.f, 0.f};
  return z;
}

// Async global -> LDS copy of 16 bytes (per lane). Tracked by ASYNCcnt.
static __device__ __forceinline__ void async_copy_b128(uint32_t lds_addr,
                                                       const void* gaddr) {
  asm volatile("global_load_async_to_lds_b128 %0, %1, off"
               :: "v"(lds_addr), "v"(gaddr) : "memory");
}

static __device__ __forceinline__ void wait_async0() {
  asm volatile("s_wait_asynccnt 0x0" ::: "memory");
}

// ---------------------------------------------------------------------------
// Kernel 0: swizzle W1 (512x64 f32) and W2 (64x16 f32) into bf16 B-fragment
// layout: fragment (kk, f), lane L holds column n = f*16 + (L&15), halves
// h=0..15 hold K = kk*32 + (L>=16 ? 8 : 0) + (h&7) + (h>=8 ? 16 : 0).
// ---------------------------------------------------------------------------
__global__ void gat_prep(const float* __restrict__ W1, const float* __restrict__ W2,
                         bf16* __restrict__ w1s, bf16* __restrict__ w2s) {
  int idx = blockIdx.x * blockDim.x + threadIdx.x;
  if (idx < 32768) {                       // W1: 16 kk * 4 f * 32 lanes * 16 halves
    int kk = idx >> 11, f = (idx >> 9) & 3, lane = (idx >> 4) & 31, h = idx & 15;
    int n = f * 16 + (lane & 15);
    int K = kk * 32 + ((lane >> 4) << 3) + (h & 7) + ((h >> 3) << 4);
    w1s[idx] = (bf16)W1[K * HID + n];
  } else if (idx < 32768 + 1024) {         // W2: 2 kk * 32 lanes * 16 halves
    int j = idx - 32768;
    int kk = j >> 9, lane = (j >> 4) & 31, h = j & 15;
    int n = lane & 15;
    int K = kk * 32 + ((lane >> 4) << 3) + (h & 7) + ((h >> 3) << 4);
    w2s[j] = (bf16)W2[K * CLS + n];
  }
}

// ---------------------------------------------------------------------------
// Kernel 1: g1 = x @ W1. 256-thread blocks (8 waves); the swizzled W1 (64 KB)
// is async-copied global->LDS once per workgroup, then every wave runs a
// 16-row tile: K=512 in 16 steps x 4 N-tiles of WMMA fed from ds_load_b128.
// Epilogue computes per-head sl/sr via shfl_xor reduction and stores g1
// transposed (feature-major) bf16 for the attention B-fragments.
// ---------------------------------------------------------------------------
__global__ __launch_bounds__(256)
void gat_proj1(const float* __restrict__ x, const float* __restrict__ a1l,
               const float* __restrict__ a1r, const bf16* __restrict__ w1s,
               bf16* __restrict__ g1t, float* __restrict__ s1lt,
               float* __restrict__ s1rt) {
  __shared__ bf16 lds_w1[32768];           // 64 KB fragment-layout copy of w1s

  const int tid  = threadIdx.x;
  const int lane = tid & 31;
  const int wave = tid >> 5;

  // Cooperative async stage-in: 4096 x 16B transfers, 16 per thread.
  {
    const uint32_t lbase = (uint32_t)(uintptr_t)(&lds_w1[0]);
#pragma unroll
    for (int it = 0; it < 16; ++it) {
      int e = (it * 256 + tid) * 8;        // element (half) index
      async_copy_b128(lbase + (uint32_t)e * 2, (const void*)(w1s + e));
    }
    wait_async0();                         // drain this wave's ASYNCcnt
  }
  __syncthreads();                         // all waves' transfers complete

  const int i0   = (blockIdx.x * 8 + wave) * 16;
  const int row  = lane & 15;
  const int koff = (lane >> 4) * 8;        // A-fragment K base offset
  const int roff = (lane >> 4) * 8;        // C-layout row-half offset

  v8f acc[4];
#pragma unroll
  for (int f = 0; f < 4; ++f) acc[f] = vzero8();

  for (int kk = 0; kk < 16; ++kk) {
    const float* xp = x + (size_t)(i0 + row) * INF + kk * 32 + koff;
    v4f x0 = *(const v4f*)(xp);
    v4f x1 = *(const v4f*)(xp + 4);
    v4f x2 = *(const v4f*)(xp + 16);
    v4f x3 = *(const v4f*)(xp + 20);
    v16bf af;
#pragma unroll
    for (int t = 0; t < 4; ++t) {
      af[t]      = (bf16)x0[t];
      af[4 + t]  = (bf16)x1[t];
      af[8 + t]  = (bf16)x2[t];
      af[12 + t] = (bf16)x3[t];
    }
#pragma unroll
    for (int f = 0; f < 4; ++f) {
      const v8bf* wp = (const v8bf*)(lds_w1 + ((size_t)(kk * 4 + f) * 32 + lane) * 16);
      v16bf bfb = combine16(wp[0], wp[1]);
      acc[f] = wmma_bf16(af, bfb, acc[f]);
    }
  }

  // Epilogue: lane holds column n = f*16 + row across rows roff..roff+7.
#pragma unroll
  for (int f = 0; f < 4; ++f) {
    int n = f * 16 + row;                  // == h*8 + d
    float wl = a1l[n];
    float wr = a1r[n];
    v8bf gv;
#pragma unroll
    for (int r = 0; r < 8; ++r) {
      float g = acc[f][r];
      gv[r] = (bf16)g;
      float pl = g * wl, pr = g * wr;
      pl += __shfl_xor(pl, 1, 32); pl += __shfl_xor(pl, 2, 32); pl += __shfl_xor(pl, 4, 32);
      pr += __shfl_xor(pr, 1, 32); pr += __shfl_xor(pr, 2, 32); pr += __shfl_xor(pr, 4, 32);
      if ((lane & 7) == 0) {
        int hh = 2 * f + ((lane >> 3) & 1);
        int rg = i0 + r + roff;
        s1lt[hh * NN + rg] = pl;
        s1rt[hh * NN + rg] = pr;
      }
    }
    *(v8bf*)(g1t + (size_t)n * NN + i0 + roff) = gv;   // 8 contiguous nodes
  }
}

// ---------------------------------------------------------------------------
// Kernel 2: layer-1 attention, flash-style. One wave per (16-row tile, head).
// p = adj ? exp(leaky(sl_i + sr_j)) : 0   (no max subtraction: |scores| << 80)
// acc += P(16x32, A-frag) @ g(32x16, B-frag, 8 valid cols), then out = acc/l,
// ELU, stored bf16 row-major for layer-2 GEMM. Next chunk is prefetched.
// ---------------------------------------------------------------------------
__global__ __launch_bounds__(32)
void gat_attn1(const unsigned char* __restrict__ adj, const bf16* __restrict__ g1t,
               const float* __restrict__ s1lt, const float* __restrict__ s1rt,
               bf16* __restrict__ h1) {
  const int lane = threadIdx.x;
  const int i0 = blockIdx.x * 16;
  const int h = blockIdx.y;
  const int row = lane & 15;
  const int koff = (lane >> 4) * 8;    // A-frag K offset
  const int kb = (lane >> 4) * 8;      // B-frag K offset
  const int d = lane & 15;             // B column (valid if < 8)
  const int dc = (d < 8) ? d : 0;
  const bool bvalid = d < 8;

  const float sl_i = s1lt[h * NN + i0 + row];
  const unsigned char* arow = adj + (size_t)(i0 + row) * NN + koff;
  const float* srow = s1rt + h * NN + koff;
  const bf16* gcol = g1t + (size_t)(h * 8 + dc) * NN + kb;

  v8f acc = vzero8();
  float lsum = 0.f;

  for (int jj = 0; jj < 128; ++jj) {
    const int jb = jj * 32;
    uint64_t m0 = *(const uint64_t*)(arow + jb);
    uint64_t m1 = *(const uint64_t*)(arow + jb + 16);
    v4f s0 = *(const v4f*)(srow + jb);
    v4f s1 = *(const v4f*)(srow + jb + 4);
    v4f s2 = *(const v4f*)(srow + jb + 16);
    v4f s3 = *(const v4f*)(srow + jb + 20);

    if (jj + 1 < 128) {                 // prefetch next chunk (speculative)
      __builtin_prefetch(arow + jb + 32, 0, 3);
      __builtin_prefetch(srow + jb + 32, 0, 3);
      __builtin_prefetch(gcol + jb + 32, 0, 3);
    }

    float ls = 0.f;
    v16bf ap;
#pragma unroll
    for (int t = 0; t < 4; ++t) {
      float e0 = sl_i + s0[t]; e0 = e0 > 0.f ? e0 : SLOPE * e0;
      float p0 = ((m0 >> (8 * t)) & 0xffu) ? __expf(e0) : 0.f;
      float e1 = sl_i + s1[t]; e1 = e1 > 0.f ? e1 : SLOPE * e1;
      float p1 = ((m0 >> (8 * (t + 4))) & 0xffu) ? __expf(e1) : 0.f;
      float e2 = sl_i + s2[t]; e2 = e2 > 0.f ? e2 : SLOPE * e2;
      float p2 = ((m1 >> (8 * t)) & 0xffu) ? __expf(e2) : 0.f;
      float e3 = sl_i + s3[t]; e3 = e3 > 0.f ? e3 : SLOPE * e3;
      float p3 = ((m1 >> (8 * (t + 4))) & 0xffu) ? __expf(e3) : 0.f;
      ls += p0 + p1 + p2 + p3;
      ap[t] = (bf16)p0; ap[4 + t] = (bf16)p1;
      ap[8 + t] = (bf16)p2; ap[12 + t] = (bf16)p3;
    }
    ls += __shfl_xor(ls, 16, 32);       // lane pair holds the same row
    lsum += ls;

    v8bf b0 = *(const v8bf*)(gcol + jb);
    v8bf b1 = *(const v8bf*)(gcol + jb + 16);
    v16bf bfb;
#pragma unroll
    for (int t = 0; t < 8; ++t) {
      bfb[t]     = bvalid ? b0[t] : (bf16)0.f;
      bfb[t + 8] = bvalid ? b1[t] : (bf16)0.f;
    }
    acc = wmma_bf16(ap, bfb, acc);
  }

  const int roff = (lane >> 4) * 8;
#pragma unroll
  for (int r = 0; r < 8; ++r) {
    int ri = r + roff;
    float l = __shfl(lsum, ri, 32);     // lanes 0..15 hold rows 0..15
    float o = acc[r] / l;
    float e = o > 0.f ? o : (__expf(o) - 1.f);   // ELU
    if (bvalid) h1[(size_t)(i0 + ri) * HID + h * 8 + d] = (bf16)e;
  }
}

// ---------------------------------------------------------------------------
// Kernel 3: g2 = h1 @ W2 (16x64 x 64x16, two WMMA steps) + s2l/s2r + g2
// transposed (class-major) bf16.
// ---------------------------------------------------------------------------
__global__ __launch_bounds__(32)
void gat_proj2(const bf16* __restrict__ h1, const float* __restrict__ a2l,
               const float* __restrict__ a2r, const bf16* __restrict__ w2s,
               bf16* __restrict__ g2t, float* __restrict__ s2l,
               float* __restrict__ s2r) {
  const int lane = threadIdx.x;
  const int i0 = blockIdx.x * 16;
  const int row = lane & 15;
  const int koff = (lane >> 4) * 8;

  v8f acc = vzero8();
#pragma unroll
  for (int kk = 0; kk < 2; ++kk) {
    const bf16* hp = h1 + (size_t)(i0 + row) * HID + kk * 32 + koff;
    v16bf af = combine16(*(const v8bf*)(hp), *(const v8bf*)(hp + 16));
    const v8bf* wp = (const v8bf*)(w2s + ((size_t)kk * 32 + lane) * 16);
    v16bf bfb = combine16(wp[0], wp[1]);
    acc = wmma_bf16(af, bfb, acc);
  }

  const int c = row;
  const float wl = a2l[c], wr = a2r[c];
  const int roff = (lane >> 4) * 8;
  v8bf gv;
#pragma unroll
  for (int r = 0; r < 8; ++r) {
    float g = acc[r];
    gv[r] = (bf16)g;
    float pl = g * wl, pr = g * wr;
    pl += __shfl_xor(pl, 1, 32); pl += __shfl_xor(pl, 2, 32);
    pl += __shfl_xor(pl, 4, 32); pl += __shfl_xor(pl, 8, 32);
    pr += __shfl_xor(pr, 1, 32); pr += __shfl_xor(pr, 2, 32);
    pr += __shfl_xor(pr, 4, 32); pr += __shfl_xor(pr, 8, 32);
    if (row == 0) {
      s2l[i0 + r + roff] = pl;
      s2r[i0 + r + roff] = pr;
    }
  }
  *(v8bf*)(g2t + (size_t)c * NN + i0 + roff) = gv;
}

// ---------------------------------------------------------------------------
// Kernel 4: layer-2 attention (1 head, 16 classes -> full 16x16 WMMA tile).
// mean over 1 head == identity. Writes final f32 output [4096, 16].
// ---------------------------------------------------------------------------
__global__ __launch_bounds__(32)
void gat_attn2(const unsigned char* __restrict__ adj, const bf16* __restrict__ g2t,
               const float* __restrict__ s2l, const float* __restrict__ s2r,
               float* __restrict__ out) {
  const int lane = threadIdx.x;
  const int i0 = blockIdx.x * 16;
  const int row = lane & 15;
  const int koff = (lane >> 4) * 8;
  const int kb = (lane >> 4) * 8;
  const int c = lane & 15;

  const float sl_i = s2l[i0 + row];
  const unsigned char* arow = adj + (size_t)(i0 + row) * NN + koff;
  const float* srow = s2r + koff;
  const bf16* gcol = g2t + (size_t)c * NN + kb;

  v8f acc = vzero8();
  float lsum = 0.f;

  for (int jj = 0; jj < 128; ++jj) {
    const int jb = jj * 32;
    uint64_t m0 = *(const uint64_t*)(arow + jb);
    uint64_t m1 = *(const uint64_t*)(arow + jb + 16);
    v4f s0 = *(const v4f*)(srow + jb);
    v4f s1 = *(const v4f*)(srow + jb + 4);
    v4f s2 = *(const v4f*)(srow + jb + 16);
    v4f s3 = *(const v4f*)(srow + jb + 20);

    if (jj + 1 < 128) {                 // prefetch next chunk (speculative)
      __builtin_prefetch(arow + jb + 32, 0, 3);
      __builtin_prefetch(srow + jb + 32, 0, 3);
      __builtin_prefetch(gcol + jb + 32, 0, 3);
    }

    float ls = 0.f;
    v16bf ap;
#pragma unroll
    for (int t = 0; t < 4; ++t) {
      float e0 = sl_i + s0[t]; e0 = e0 > 0.f ? e0 : SLOPE * e0;
      float p0 = ((m0 >> (8 * t)) & 0xffu) ? __expf(e0) : 0.f;
      float e1 = sl_i + s1[t]; e1 = e1 > 0.f ? e1 : SLOPE * e1;
      float p1 = ((m0 >> (8 * (t + 4))) & 0xffu) ? __expf(e1) : 0.f;
      float e2 = sl_i + s2[t]; e2 = e2 > 0.f ? e2 : SLOPE * e2;
      float p2 = ((m1 >> (8 * t)) & 0xffu) ? __expf(e2) : 0.f;
      float e3 = sl_i + s3[t]; e3 = e3 > 0.f ? e3 : SLOPE * e3;
      float p3 = ((m1 >> (8 * (t + 4))) & 0xffu) ? __expf(e3) : 0.f;
      ls += p0 + p1 + p2 + p3;
      ap[t] = (bf16)p0; ap[4 + t] = (bf16)p1;
      ap[8 + t] = (bf16)p2; ap[12 + t] = (bf16)p3;
    }
    ls += __shfl_xor(ls, 16, 32);
    lsum += ls;

    v16bf bfb = combine16(*(const v8bf*)(gcol + jb), *(const v8bf*)(gcol + jb + 16));
    acc = wmma_bf16(ap, bfb, acc);
  }

  const int roff = (lane >> 4) * 8;
#pragma unroll
  for (int r = 0; r < 8; ++r) {
    int ri = r + roff;
    float l = __shfl(lsum, ri, 32);
    out[(size_t)(i0 + ri) * CLS + c] = acc[r] / l;
  }
}

// ---------------------------------------------------------------------------
extern "C" void kernel_launch(void* const* d_in, const int* in_sizes, int n_in,
                              void* d_out, int out_size, void* d_ws, size_t ws_size,
                              hipStream_t stream) {
  (void)in_sizes; (void)n_in; (void)out_size; (void)ws_size;

  const float* x           = (const float*)d_in[0];
  const unsigned char* adj = (const unsigned char*)d_in[1];  // JAX bool: 1 byte/elem
  const float* W1  = (const float*)d_in[2];
  const float* a1l = (const float*)d_in[3];
  const float* a1r = (const float*)d_in[4];
  const float* W2  = (const float*)d_in[5];
  const float* a2l = (const float*)d_in[6];
  const float* a2r = (const float*)d_in[7];
  float* out = (float*)d_out;

  char* ws = (char*)d_ws;
  bf16*  w1s  = (bf16*)(ws + OFF_W1S);
  bf16*  w2s  = (bf16*)(ws + OFF_W2S);
  bf16*  g1t  = (bf16*)(ws + OFF_G1T);
  float* s1lt = (float*)(ws + OFF_S1L);
  float* s1rt = (float*)(ws + OFF_S1R);
  bf16*  h1   = (bf16*)(ws + OFF_H1);
  bf16*  g2t  = (bf16*)(ws + OFF_G2T);
  float* s2l  = (float*)(ws + OFF_S2L);
  float* s2r  = (float*)(ws + OFF_S2R);

  gat_prep <<<132, 256, 0, stream>>>(W1, W2, w1s, w2s);
  gat_proj1<<<NN / (16 * 8), 256, 0, stream>>>(x, a1l, a1r, w1s, g1t, s1lt, s1rt);
  gat_attn1<<<dim3(NN / 16, HEADS), 32, 0, stream>>>(adj, g1t, s1lt, s1rt, h1);
  gat_proj2<<<NN / 16, 32, 0, stream>>>(h1, a2l, a2r, w2s, g2t, s2l, s2r);
  gat_attn2<<<NN / 16, 32, 0, stream>>>(adj, g2t, s2l, s2r, out);
}
